// MultiHeadAttention_34299608826647
// MI455X (gfx1250) — matrix operand order repaired and checked by reading, not compiled, not run on hardware
//
#include <hip/hip_runtime.h>
#include <math.h>

// ---------- types ----------
typedef __attribute__((ext_vector_type(16))) __bf16        v16bf;
typedef __attribute__((ext_vector_type(2)))  __bf16        v2bf;
typedef __attribute__((ext_vector_type(8)))  float         v8f;
typedef __attribute__((ext_vector_type(4)))  unsigned int  v4u;

union Frag {
  v4u   q[2];   // 32 bytes = 16 bf16
  v16bf v;
};

// ---------- fp32 -> bf16 (RNE). Prefer HW v_cvt_pk_bf16_f32 ----------
#if __has_builtin(__builtin_amdgcn_cvt_pk_bf16_f32)
__device__ __forceinline__ unsigned int pack2(float lo, float hi) {
  v2bf r = __builtin_amdgcn_cvt_pk_bf16_f32(lo, hi);
  return __builtin_bit_cast(unsigned int, r);
}
__device__ __forceinline__ unsigned short f2bf(float f) {
  return (unsigned short)(pack2(f, 0.f) & 0xffffu);
}
#else
__device__ __forceinline__ unsigned short f2bf(float f) {
  unsigned int u = __float_as_uint(f);
  u += 0x7FFFu + ((u >> 16) & 1u);   // round to nearest even
  return (unsigned short)(u >> 16);
}
__device__ __forceinline__ unsigned int pack2(float lo, float hi) {
  return (unsigned int)f2bf(lo) | ((unsigned int)f2bf(hi) << 16);
}
#endif

__device__ __forceinline__ v8f wmma_bf16(v16bf a, v16bf b, v8f c) {
  // (neg_a, A, neg_b, B, c_mod, C, reuse_a, reuse_b)
  return __builtin_amdgcn_wmma_f32_16x16x32_bf16(false, a, false, b, (short)0, c,
                                                 false, false);
}

// ---------- DPP16 butterfly exchange (stays within a 16-lane row) ----------
template <int CTRL>
__device__ __forceinline__ float dpp_xchg(float x) {
  int r = __builtin_amdgcn_update_dpp(0, __float_as_int(x), CTRL, 0xf, 0xf, true);
  return __int_as_float(r);
}
// Reduce over the 16 lanes of each half-wave; every lane gets the result.
__device__ __forceinline__ float half_reduce_max(float x) {
  x = fmaxf(x, dpp_xchg<0xB1>(x));   // quad_perm [1,0,3,2]  (xor 1)
  x = fmaxf(x, dpp_xchg<0x4E>(x));   // quad_perm [2,3,0,1]  (xor 2)
  x = fmaxf(x, dpp_xchg<0x141>(x));  // row_half_mirror      (~xor 4)
  x = fmaxf(x, dpp_xchg<0x140>(x));  // row_mirror           (~xor 8)
  return x;
}
__device__ __forceinline__ float half_reduce_sum(float x) {
  x += dpp_xchg<0xB1>(x);
  x += dpp_xchg<0x4E>(x);
  x += dpp_xchg<0x141>(x);
  x += dpp_xchg<0x140>(x);
  return x;
}

// =====================================================================
// GEMM: Out[m][n] = sum_k X[m][k] * W[n][k] + bias[n]
// M=4096, N=K=1024.  Block tile 128x128, 8 waves, BK=32.
// X_BF16: X is bf16 (workspace) else fp32.  OUT_F32: store fp32 (d_out).
// HEADT:  store transposed per head -> Vt[b][h][dh][s] (bf16).
// =====================================================================
template <bool X_BF16, bool OUT_F32, bool HEADT>
__global__ __launch_bounds__(256) void gemm_bias_kernel(
    const void* __restrict__ Xv, const float* __restrict__ W,
    const float* __restrict__ bias, void* __restrict__ Outv) {
  __shared__ __align__(16) unsigned int As[128 * 20];  // 128 rows x 40 bf16 (pad)
  __shared__ __align__(16) unsigned int Bs[128 * 20];

  const int tid  = threadIdx.x;
  const int lane = tid & 31;
  const int wave = tid >> 5;
  const int ln   = lane & 15;
  const int hl   = lane >> 4;
  const int m_off = (wave & 3) * 32;   // 4 waves down
  const int n_off = (wave >> 2) * 64;  // 2 waves across
  const int bm0 = blockIdx.y * 128;
  const int bn0 = blockIdx.x * 128;

  const int ldr = tid >> 1;  // 0..127
  const int sg  = tid & 1;   // 16-element k segment

  v8f zero = {};
  v8f acc[2][4];
#pragma unroll
  for (int i = 0; i < 2; ++i)
#pragma unroll
    for (int j = 0; j < 4; ++j) acc[i][j] = zero;

  for (int k0 = 0; k0 < 1024; k0 += 32) {
    {
      unsigned int* dst = &As[ldr * 20 + sg * 8];
      if (X_BF16) {
        const unsigned short* src =
            (const unsigned short*)Xv + (size_t)(bm0 + ldr) * 1024 + k0 + sg * 16;
        ((v4u*)dst)[0] = *(const v4u*)src;
        ((v4u*)dst)[1] = *(const v4u*)(src + 8);
      } else {
        const float* src =
            (const float*)Xv + (size_t)(bm0 + ldr) * 1024 + k0 + sg * 16;
#pragma unroll
        for (int i = 0; i < 8; ++i) dst[i] = pack2(src[2 * i], src[2 * i + 1]);
      }
      const float* srcB = W + (size_t)(bn0 + ldr) * 1024 + k0 + sg * 16;
      unsigned int* dstB = &Bs[ldr * 20 + sg * 8];
#pragma unroll
      for (int i = 0; i < 8; ++i) dstB[i] = pack2(srcB[2 * i], srcB[2 * i + 1]);
    }
    __syncthreads();

    // A fragments: lane(r,hl) holds row r, k = {8hl..8hl+7, 16+8hl..16+8hl+7}
    Frag a[2];
#pragma unroll
    for (int mt = 0; mt < 2; ++mt) {
      const unsigned int* p = &As[(m_off + mt * 16 + ln) * 20 + hl * 4];
      a[mt].q[0] = *(const v4u*)p;
      a[mt].q[1] = *(const v4u*)(p + 8);
    }
    // B fragments: lane(n,hl) holds W-row n, k = {8hl..8hl+7, 16+8hl..16+8hl+7}
#pragma unroll
    for (int nt = 0; nt < 4; ++nt) {
      Frag bf;
      const unsigned int* p = &Bs[(n_off + nt * 16 + ln) * 20 + hl * 4];
      bf.q[0] = *(const v4u*)p;
      bf.q[1] = *(const v4u*)(p + 8);
#pragma unroll
      for (int mt = 0; mt < 2; ++mt)
        acc[mt][nt] = wmma_bf16(a[mt].v, bf.v, acc[mt][nt]);
    }
    __syncthreads();
  }

  // Epilogue: C layout -> lane(n,hl), element j = row (j + 8*hl)
#pragma unroll
  for (int mt = 0; mt < 2; ++mt) {
#pragma unroll
    for (int nt = 0; nt < 4; ++nt) {
      const int n_g = bn0 + n_off + nt * 16 + ln;
      const float bv = bias[n_g];
#pragma unroll
      for (int j = 0; j < 8; ++j) {
        const int m_g = bm0 + m_off + mt * 16 + j + 8 * hl;
        const float val = acc[mt][nt][j] + bv;
        if (HEADT) {
          const int hh = n_g >> 6, dh = n_g & 63;
          const int bb = m_g >> 10, ss = m_g & 1023;
          ((unsigned short*)Outv)[(((size_t)(bb * 16 + hh) * 64 + dh) << 10) + ss] =
              f2bf(val);
        } else if (OUT_F32) {
          ((float*)Outv)[(size_t)m_g * 1024 + n_g] = val;
        } else {
          ((unsigned short*)Outv)[(size_t)m_g * 1024 + n_g] = f2bf(val);
        }
      }
    }
  }
}

// =====================================================================
// Flash-style attention. Block = (b, h, 128 queries). 8 waves x 16 q.
// Streams keys in chunks of 64 with online softmax; P round-trips
// through per-wave LDS to become a WMMA A-operand. Softmax reductions
// are pure-VALU DPP16 butterflies (no LDS traffic).
// =====================================================================
__global__ __launch_bounds__(256) void attn_kernel(
    const unsigned short* __restrict__ Qb, const unsigned short* __restrict__ Kb,
    const unsigned short* __restrict__ Vt, const int* __restrict__ vls,
    unsigned short* __restrict__ C2) {
  __shared__ __align__(16) unsigned short P[8][16 * 72];  // 72 pitch: bank-friendly

  const int b = blockIdx.z, h = blockIdx.y;
  const int wave = threadIdx.x >> 5;
  const int lane = threadIdx.x & 31;
  const int ln = lane & 15, hl = lane >> 4;
  const int q0 = blockIdx.x * 128 + wave * 16;
  const int vlen = vls[b];
  unsigned short* Pw = P[wave];

  // Q fragments (A operand), dh chunks {0..31},{32..63}
  Frag aq[2];
  {
    const unsigned short* qp =
        Qb + (size_t)(b * 1024 + q0 + ln) * 1024 + h * 64;
#pragma unroll
    for (int c = 0; c < 2; ++c) {
      const unsigned short* p = qp + c * 32 + 16 * hl;
      aq[c].q[0] = *(const v4u*)p;
      aq[c].q[1] = *(const v4u*)(p + 8);
    }
  }

  v8f zero = {};
  float mrow[8], lrow[8];
  v8f O[4];
#pragma unroll
  for (int j = 0; j < 8; ++j) { mrow[j] = -INFINITY; lrow[j] = 0.f; }
#pragma unroll
  for (int t = 0; t < 4; ++t) O[t] = zero;

  const size_t kbase = (size_t)(b * 1024) * 1024 + h * 64;
  const size_t vbase = (size_t)(b * 16 + h) * 64 * 1024;

  for (int kc = 0; kc < 1024; kc += 64) {
    // ---- scores for 64-key chunk: 4 tiles of 16 keys
    v8f sc[4];
#pragma unroll
    for (int t = 0; t < 4; ++t) {
      const unsigned short* kp = Kb + kbase + (size_t)(kc + t * 16 + ln) * 1024;
      Frag kf0, kf1;  // B operand: lane = key, elements = dh
      kf0.q[0] = *(const v4u*)(kp + 16 * hl);
      kf0.q[1] = *(const v4u*)(kp + 16 * hl + 8);
      kf1.q[0] = *(const v4u*)(kp + 32 + 16 * hl);
      kf1.q[1] = *(const v4u*)(kp + 32 + 16 * hl + 8);
      v8f s = zero;
      s = wmma_bf16(aq[0].v, kf0.v, s);
      s = wmma_bf16(aq[1].v, kf1.v, s);
      sc[t] = s;
    }
    if (kc + 64 < 1024) {
      __builtin_prefetch(Kb + kbase + (size_t)(kc + 64 + ln) * 1024, 0, 0);
      __builtin_prefetch(Vt + vbase + (size_t)ln * 1024 + kc + 64, 0, 0);
    }

    // ---- scale, pad-mask (exact -1e6 like reference), chunk max
    float cmax[8];
#pragma unroll
    for (int j = 0; j < 8; ++j) cmax[j] = -INFINITY;
#pragma unroll
    for (int t = 0; t < 4; ++t) {
      const int col = kc + t * 16 + ln;
      const bool msk = col >= vlen;
#pragma unroll
      for (int j = 0; j < 8; ++j) {
        float v = msk ? -1.0e6f : sc[t][j] * 0.125f;
        sc[t][j] = v;
        cmax[j] = fmaxf(cmax[j], v);
      }
    }
    // ---- online softmax; DPP16 reductions stay within each lane half,
    //      which matches rows j (lanes 0-15) and j+8 (lanes 16-31).
#pragma unroll
    for (int j = 0; j < 8; ++j) {
      cmax[j] = half_reduce_max(cmax[j]);
      const float mn = fmaxf(mrow[j], cmax[j]);
      const float so = __expf(mrow[j] - mn);
      mrow[j] = mn;
      float rs = 0.f;
#pragma unroll
      for (int t = 0; t < 4; ++t) {
        const float p = __expf(sc[t][j] - mn);
        sc[t][j] = p;
        rs += p;
      }
      rs = half_reduce_sum(rs);
      lrow[j] = lrow[j] * so + rs;
#pragma unroll
      for (int t = 0; t < 4; ++t) O[t][j] *= so;
    }

    // ---- P (C layout) -> LDS -> reload as A operand (same-wave DS is in-order)
#pragma unroll
    for (int t = 0; t < 4; ++t)
#pragma unroll
      for (int j = 0; j < 8; ++j)
        Pw[(j + 8 * hl) * 72 + t * 16 + ln] = f2bf(sc[t][j]);

    Frag pa[2];
#pragma unroll
    for (int c = 0; c < 2; ++c) {
      const unsigned short* p = Pw + ln * 72 + c * 32 + 16 * hl;
      pa[c].q[0] = *(const v4u*)p;
      pa[c].q[1] = *(const v4u*)(p + 8);
    }

    // ---- O += P @ V  (Vt rows contiguous in key dim)
#pragma unroll
    for (int dt = 0; dt < 4; ++dt) {
      const unsigned short* vp = Vt + vbase + (size_t)(dt * 16 + ln) * 1024 + kc;
      Frag v0, v1;
      v0.q[0] = *(const v4u*)(vp + 16 * hl);
      v0.q[1] = *(const v4u*)(vp + 16 * hl + 8);
      v1.q[0] = *(const v4u*)(vp + 32 + 16 * hl);
      v1.q[1] = *(const v4u*)(vp + 32 + 16 * hl + 8);
      O[dt] = wmma_bf16(pa[0].v, v0.v, O[dt]);
      O[dt] = wmma_bf16(pa[1].v, v1.v, O[dt]);
    }
  }

  // ---- normalize, store merged-head context (bf16)
#pragma unroll
  for (int dt = 0; dt < 4; ++dt)
#pragma unroll
    for (int j = 0; j < 8; ++j) {
      const float val = O[dt][j] / lrow[j];
      C2[(size_t)(b * 1024 + q0 + j + 8 * hl) * 1024 + h * 64 + dt * 16 + ln] =
          f2bf(val);
    }
}

// =====================================================================
extern "C" void kernel_launch(void* const* d_in, const int* in_sizes, int n_in,
                              void* d_out, int out_size, void* d_ws, size_t ws_size,
                              hipStream_t stream) {
  (void)in_sizes; (void)n_in; (void)out_size; (void)ws_size;
  const float* queries = (const float*)d_in[0];
  const float* keys    = (const float*)d_in[1];
  const float* values  = (const float*)d_in[2];
  const int*   vls     = (const int*)d_in[3];
  const float* W_q = (const float*)d_in[4];
  const float* b_q = (const float*)d_in[5];
  const float* W_k = (const float*)d_in[6];
  const float* b_k = (const float*)d_in[7];
  const float* W_v = (const float*)d_in[8];
  const float* b_v = (const float*)d_in[9];
  const float* W_o = (const float*)d_in[10];
  const float* b_o = (const float*)d_in[11];

  char* ws = (char*)d_ws;                       // 4 x 8MB bf16 buffers
  unsigned short* Qb = (unsigned short*)(ws);
  unsigned short* Kb = (unsigned short*)(ws + ((size_t)8 << 20));
  unsigned short* Vt = (unsigned short*)(ws + ((size_t)16 << 20));
  unsigned short* C2 = (unsigned short*)(ws + ((size_t)24 << 20));

  dim3 gg(8, 32, 1);  // N/128, M/128
  gemm_bias_kernel<false, false, false><<<gg, 256, 0, stream>>>(queries, W_q, b_q, Qb);
  gemm_bias_kernel<false, false, false><<<gg, 256, 0, stream>>>(keys,    W_k, b_k, Kb);
  gemm_bias_kernel<false, false, true ><<<gg, 256, 0, stream>>>(values,  W_v, b_v, Vt);
  attn_kernel<<<dim3(8, 16, 4), 256, 0, stream>>>(Qb, Kb, Vt, vls, C2);
  gemm_bias_kernel<true, true, false><<<gg, 256, 0, stream>>>(C2, W_o, b_o, d_out);
}
